// CLIPLoss_10599979286745
// MI455X (gfx1250) — compile-verified, matrix operand order checked by reading, not run on hardware
//
#include <hip/hip_runtime.h>
#include <hip/hip_bf16.h>

typedef __attribute__((ext_vector_type(16))) __bf16    v16bf;
typedef __attribute__((ext_vector_type(8)))  float     v8f;
typedef __attribute__((ext_vector_type(8)))  unsigned  v8u;
typedef __attribute__((ext_vector_type(4)))  unsigned  u32x4;
typedef __attribute__((ext_vector_type(8)))  int       i32x8;
typedef __attribute__((ext_vector_type(4)))  int       i32x4;

#define LDSK 264   // padded bf16 row stride (256 data + 8 pad) -> breaks LDS bank conflicts

#if __has_builtin(__builtin_amdgcn_tensor_load_to_lds) && __has_builtin(__builtin_amdgcn_s_wait_tensorcnt)
#define HAVE_TDM 1
#else
#define HAVE_TDM 0
#endif

union FragU { v8u u; v16bf h; };

__device__ __forceinline__ unsigned bf16h(float x) {
    unsigned u = __float_as_uint(x);
    return (u + 0x7FFFu + ((u >> 16) & 1u)) >> 16;   // RNE f32 -> bf16 bits
}
__device__ __forceinline__ float bf16f(unsigned h) { return __uint_as_float(h << 16); }

// split two floats into packed bf16 hi dword + bf16 lo (residual) dword
__device__ __forceinline__ void cvt2(float x0, float x1, unsigned& hi, unsigned& lo) {
    unsigned h0 = bf16h(x0), h1 = bf16h(x1);
    unsigned l0 = bf16h(x0 - bf16f(h0)), l1 = bf16h(x1 - bf16f(h1));
    hi = h0 | (h1 << 16);
    lo = l0 | (l1 << 16);
}

// load a 16x32 bf16 operand fragment (ISA A/B lane layout) from any 16B-aligned base
__device__ __forceinline__ void load_frag(const unsigned short* base, FragU& F) {
    const uint4 b0 = *reinterpret_cast<const uint4*>(base);
    const uint4 b1 = *reinterpret_cast<const uint4*>(base + 16);
    F.u[0] = b0.x; F.u[1] = b0.y; F.u[2] = b0.z; F.u[3] = b0.w;
    F.u[4] = b1.x; F.u[5] = b1.y; F.u[6] = b1.z; F.u[7] = b1.w;
}

#if HAVE_TDM
// TDM: DMA a 128x256 bf16 tile (row-major, 512B rows) global -> LDS, inserting
// 16B of LDS padding after every row (matches LDSK=264). D# per ISA 08_async_tensor.
// This toolchain's builtin takes 6 args: (g0 u32x4, g1 i32x8, g2 i32x4, g3 i32x4, i32x8, cpol)
__device__ __forceinline__ void tdm_load_wtile(const unsigned short* gsrc, unsigned lds_addr) {
    const unsigned long long ga = (unsigned long long)(uintptr_t)gsrc;
    u32x4 g0;
    g0[0] = 1u;                                        // count=1, user mode
    g0[1] = lds_addr;                                  // LDS byte address
    g0[2] = (unsigned)(ga & 0xFFFFFFFFu);              // global_addr[31:0]
    g0[3] = (unsigned)((ga >> 32) & 0x01FFFFFFu)       // global_addr[56:32]
          | (2u << 30);                                // type=2 ("image")
    i32x8 g1;
    g1[0] = (1 << 16)        // data_size = 2 bytes (bf16)
          | (1 << 20)        // pad_enable
          | (6 << 22)        // pad_interval: 2^(6+1)=128 DWORDs (one 512B row)
          | (3 << 25);       // pad_amount: 4 DWORDs (16B) -> row stride 528B
    g1[1] = (int)(256u << 16);   // tensor_dim0 = 256 elems (low16 @ [31:16])
    g1[2] = (int)(128u << 16);   // tensor_dim0 hi16=0; tensor_dim1 = 128 (low16 @ [31:16])
    g1[3] = (int)(256u << 16);   // tensor_dim1 hi16=0; tile_dim0 = 256
    g1[4] = 128;                 // tile_dim1 = 128; tile_dim2 = 0
    g1[5] = 256;                 // tensor_dim0_stride = 256 elems (low32)
    g1[6] = 0;                   // stride hi bits / tensor_dim1_stride (unused, 2D)
    g1[7] = 0;
    const i32x4 z4 = {0, 0, 0, 0};
    const i32x8 z8 = {0, 0, 0, 0, 0, 0, 0, 0};
    __builtin_amdgcn_tensor_load_to_lds(g0, g1, z4, z4, z8, 0);
}
#endif

// ---------------------------------------------------------------------------
// Pre-pass: convert V and W (f32) to bf16 hi/lo arrays ONCE (64x fewer
// conversions than doing it per (i,j) block). 4 elements / thread.
// ---------------------------------------------------------------------------
__global__ __launch_bounds__(256) void precvt_kernel(const float* __restrict__ V,
                                                     const float* __restrict__ W,
                                                     unsigned short* __restrict__ Vh,
                                                     unsigned short* __restrict__ Vl,
                                                     unsigned short* __restrict__ Wh,
                                                     unsigned short* __restrict__ Wl) {
    const size_t N = (size_t)64 * 128 * 256;
    size_t e = ((size_t)blockIdx.x * 256 + threadIdx.x) * 4;
    const float* src;
    unsigned short *dh, *dl;
    if (e < N) { src = V; dh = Vh; dl = Vl; }
    else       { src = W; dh = Wh; dl = Wl; e -= N; }
    const float4 f = *reinterpret_cast<const float4*>(src + e);
    unsigned h01, l01, h23, l23;
    cvt2(f.x, f.y, h01, l01);
    cvt2(f.z, f.w, h23, l23);
    *reinterpret_cast<uint2*>(dh + e) = make_uint2(h01, h23);
    *reinterpret_cast<uint2*>(dl + e) = make_uint2(l01, l23);
}

// ---------------------------------------------------------------------------
// Shared reduction pipeline.
// acc tile layout: reg r, lanes 0-15 -> M=16w+r, N=lane; lanes 16-31 -> M=16w+r+8
// ---------------------------------------------------------------------------
struct SmemRed {
    float sPA[8 * 128];
    float sPB[8 * 128];
    float sCM[128];
    float sSV[128];
    float sSW[128];
};

__device__ __forceinline__ void softpool_reduce(v8f acc[8], SmemRed* sm,
                                                int tid, int wave, int lane,
                                                int l15, int halfsel,
                                                float* __restrict__ dense, int p) {
    // ---- v2w: softmax over N within each row (tau=0.5 -> *2 in exponent) ----
#pragma unroll
    for (int r = 0; r < 8; ++r) {
        float m = acc[0][r];
#pragma unroll
        for (int nb = 1; nb < 8; ++nb) m = fmaxf(m, acc[nb][r]);
        for (int mask = 1; mask <= 8; mask <<= 1) m = fmaxf(m, __shfl_xor(m, mask, 32));
        float z = 0.f, num = 0.f;
#pragma unroll
        for (int nb = 0; nb < 8; ++nb) {
            const float s = acc[nb][r];
            const float e = __expf((s - m) * 2.0f);
            z += e; num += s * e;
        }
        for (int mask = 1; mask <= 8; mask <<= 1) {
            z += __shfl_xor(z, mask, 32);
            num += __shfl_xor(num, mask, 32);
        }
        if (l15 == 0) sm->sSV[(wave << 4) + r + (halfsel ? 8 : 0)] = num / z;
    }

    // ---- w2v: softmax over M per column, cross-wave via LDS ----
#pragma unroll
    for (int nb = 0; nb < 8; ++nb) {
        float pm = acc[nb][0];
#pragma unroll
        for (int r = 1; r < 8; ++r) pm = fmaxf(pm, acc[nb][r]);
        pm = fmaxf(pm, __shfl_xor(pm, 16, 32));
        if (lane < 16) sm->sPA[wave * 128 + (nb << 4) + lane] = pm;
    }
    __syncthreads();
    if (tid < 128) {
        float cm = sm->sPA[tid];
        for (int w = 1; w < 8; ++w) cm = fmaxf(cm, sm->sPA[w * 128 + tid]);
        sm->sCM[tid] = cm;
    }
    __syncthreads();
#pragma unroll
    for (int nb = 0; nb < 8; ++nb) {
        const float cm = sm->sCM[(nb << 4) + l15];
        float z = 0.f, num = 0.f;
#pragma unroll
        for (int r = 0; r < 8; ++r) {
            const float s = acc[nb][r];
            const float e = __expf((s - cm) * 2.0f);
            z += e; num += s * e;
        }
        z   += __shfl_xor(z, 16, 32);
        num += __shfl_xor(num, 16, 32);
        if (lane < 16) {
            sm->sPA[wave * 128 + (nb << 4) + lane] = z;
            sm->sPB[wave * 128 + (nb << 4) + lane] = num;
        }
    }
    __syncthreads();
    if (tid < 128) {
        float z = 0.f, num = 0.f;
        for (int w = 0; w < 8; ++w) { z += sm->sPA[w * 128 + tid]; num += sm->sPB[w * 128 + tid]; }
        sm->sSW[tid] = num / z;
    }
    __syncthreads();

    // ---- frame-level softmax pooling -> dense[i,j] ----
    if (wave == 0) {
        float g[2];
#pragma unroll
        for (int q = 0; q < 2; ++q) {
            const float* s = q ? sm->sSW : sm->sSV;
            const float a0 = s[lane], a1 = s[lane + 32], a2 = s[lane + 64], a3 = s[lane + 96];
            float m = fmaxf(fmaxf(a0, a1), fmaxf(a2, a3));
            for (int mask = 1; mask <= 16; mask <<= 1) m = fmaxf(m, __shfl_xor(m, mask, 32));
            const float e0 = __expf((a0 - m) * 2.f), e1 = __expf((a1 - m) * 2.f);
            const float e2 = __expf((a2 - m) * 2.f), e3 = __expf((a3 - m) * 2.f);
            float z   = e0 + e1 + e2 + e3;
            float num = a0 * e0 + a1 * e1 + a2 * e2 + a3 * e3;
            for (int mask = 1; mask <= 16; mask <<= 1) {
                z += __shfl_xor(z, mask, 32);
                num += __shfl_xor(num, mask, 32);
            }
            g[q] = num / z;
        }
        if (lane == 0) dense[p] = 0.5f * (g[0] + g[1]);
    }
}

// interleaved dual-tile WMMA step: breaks RAW chains on the accumulators
#define WMMA3X2(accA, accB, Ah, Al, Bh0, Bl0, Bh1, Bl1)                                   \
    accA = __builtin_amdgcn_wmma_f32_16x16x32_bf16(false, Ah.h, false, Bh0.h, (short)0,   \
                                                   accA, false, false);                   \
    accB = __builtin_amdgcn_wmma_f32_16x16x32_bf16(false, Ah.h, false, Bh1.h, (short)0,   \
                                                   accB, false, false);                   \
    accA = __builtin_amdgcn_wmma_f32_16x16x32_bf16(false, Ah.h, false, Bl0.h, (short)0,   \
                                                   accA, false, false);                   \
    accB = __builtin_amdgcn_wmma_f32_16x16x32_bf16(false, Ah.h, false, Bl1.h, (short)0,   \
                                                   accB, false, false);                   \
    accA = __builtin_amdgcn_wmma_f32_16x16x32_bf16(false, Al.h, false, Bh0.h, (short)0,   \
                                                   accA, false, false);                   \
    accB = __builtin_amdgcn_wmma_f32_16x16x32_bf16(false, Al.h, false, Bh1.h, (short)0,   \
                                                   accB, false, false)

// ---------------------------------------------------------------------------
// Main fused kernel (pre-converted bf16 path): one block per (i,j) pair.
// ---------------------------------------------------------------------------
__global__ __launch_bounds__(256) void clip_pair_pre(const unsigned short* __restrict__ Vh,
                                                     const unsigned short* __restrict__ Vl,
                                                     const unsigned short* __restrict__ Wh,
                                                     const unsigned short* __restrict__ Wl,
                                                     float* __restrict__ dense) {
    __shared__ unsigned short sWh[128 * LDSK];
    __shared__ unsigned short sWl[128 * LDSK];
    __shared__ SmemRed red;

    const int p    = blockIdx.x;
    const int bi   = p >> 6;
    const int bj   = p & 63;
    const int tid  = threadIdx.x;
    const int wave = tid >> 5;
    const int lane = tid & 31;
    const int l15  = lane & 15;
    const int halfsel = (lane >> 4) & 1;

    const unsigned short* WhG = Wh + (size_t)bj * (128 * 256);
    const unsigned short* WlG = Wl + (size_t)bj * (128 * 256);

#if HAVE_TDM
    // ---- stage W_j hi/lo via Tensor Data Mover (async DMA + LDS row padding) ----
    if (wave == 0) {
        tdm_load_wtile(WhG, (unsigned)(uintptr_t)&sWh[0]);
        tdm_load_wtile(WlG, (unsigned)(uintptr_t)&sWl[0]);
        __builtin_amdgcn_s_wait_tensorcnt(0);
    }
    __syncthreads();
#else
    // ---- manual b128 staging copies ----
    for (int c = tid; c < 128 * 32; c += 256) {
        const int row = c >> 5, seg = (c & 31) << 3;
        *reinterpret_cast<uint4*>(&sWh[row * LDSK + seg]) =
            *reinterpret_cast<const uint4*>(WhG + row * 256 + seg);
        *reinterpret_cast<uint4*>(&sWl[row * LDSK + seg]) =
            *reinterpret_cast<const uint4*>(WlG + row * 256 + seg);
    }
    __syncthreads();
#endif

    // ---- GEMM: A fragments straight from global bf16 (L2-resident) ----
    const int mrow = (wave << 4) + l15;
    const unsigned short* VhP = Vh + ((size_t)bi * 128 + mrow) * 256;
    const unsigned short* VlP = Vl + ((size_t)bi * 128 + mrow) * 256;

    const v8f vzero = {0.f, 0.f, 0.f, 0.f, 0.f, 0.f, 0.f, 0.f};
    v8f acc[8];
#pragma unroll
    for (int nb = 0; nb < 8; ++nb) acc[nb] = vzero;

#pragma unroll
    for (int kc = 0; kc < 8; ++kc) {
        const int k0 = kc * 32 + halfsel * 8;   // 16-bit A layout: lanes<16 K0..7/16..23, lanes>=16 +8
        FragU Ah, Al;
        load_frag(VhP + k0, Ah);
        load_frag(VlP + k0, Al);
#pragma unroll
        for (int nb = 0; nb < 8; nb += 2) {
            const int rb0 = (nb << 4) + l15;
            const int rb1 = rb0 + 16;
            FragU Bh0, Bl0, Bh1, Bl1;
            load_frag(&sWh[rb0 * LDSK + k0], Bh0);
            load_frag(&sWl[rb0 * LDSK + k0], Bl0);
            load_frag(&sWh[rb1 * LDSK + k0], Bh1);
            load_frag(&sWl[rb1 * LDSK + k0], Bl1);
            // error-compensated bf16: VhWh + VhWl + VlWh (~fp32 accuracy), 2 tiles interleaved
            WMMA3X2(acc[nb], acc[nb + 1], Ah, Al, Bh0, Bl0, Bh1, Bl1);
        }
    }

    softpool_reduce(acc, &red, tid, wave, lane, l15, halfsel, dense, p);
}

// ---------------------------------------------------------------------------
// Fallback (small ws): in-kernel fp32 -> bf16 hi/lo conversion.
// ---------------------------------------------------------------------------
__global__ __launch_bounds__(256) void clip_pair_f32(const float* __restrict__ V,
                                                     const float* __restrict__ W,
                                                     float* __restrict__ dense) {
    __shared__ unsigned short sWh[128 * LDSK];
    __shared__ unsigned short sWl[128 * LDSK];
    __shared__ SmemRed red;

    const int p    = blockIdx.x;
    const int bi   = p >> 6;
    const int bj   = p & 63;
    const int tid  = threadIdx.x;
    const int wave = tid >> 5;
    const int lane = tid & 31;
    const int l15  = lane & 15;
    const int halfsel = (lane >> 4) & 1;

    const float* Wj = W + (size_t)bj * (128 * 256);
    for (int e = tid * 4; e < 128 * 256; e += 256 * 4) {
        const int row = e >> 8, col = e & 255;
        const float4 f = *reinterpret_cast<const float4*>(Wj + e);
        unsigned h01, l01, h23, l23;
        cvt2(f.x, f.y, h01, l01);
        cvt2(f.z, f.w, h23, l23);
        unsigned* ph = reinterpret_cast<unsigned*>(&sWh[row * LDSK + col]);
        unsigned* pl = reinterpret_cast<unsigned*>(&sWl[row * LDSK + col]);
        ph[0] = h01; ph[1] = h23;
        pl[0] = l01; pl[1] = l23;
    }
    __syncthreads();

    const int mrow = (wave << 4) + l15;
    const float* Vp = V + ((size_t)bi * 128 + mrow) * 256;

    const v8f vzero = {0.f, 0.f, 0.f, 0.f, 0.f, 0.f, 0.f, 0.f};
    v8f acc[8];
#pragma unroll
    for (int nb = 0; nb < 8; ++nb) acc[nb] = vzero;

#pragma unroll
    for (int kc = 0; kc < 8; ++kc) {
        const int k0 = kc * 32 + halfsel * 8;
        FragU Ah, Al;
        {
            const float4 a0 = *reinterpret_cast<const float4*>(Vp + k0);
            const float4 a1 = *reinterpret_cast<const float4*>(Vp + k0 + 4);
            const float4 a2 = *reinterpret_cast<const float4*>(Vp + k0 + 16);
            const float4 a3 = *reinterpret_cast<const float4*>(Vp + k0 + 20);
            unsigned h, l;
            cvt2(a0.x, a0.y, h, l); Ah.u[0] = h; Al.u[0] = l;
            cvt2(a0.z, a0.w, h, l); Ah.u[1] = h; Al.u[1] = l;
            cvt2(a1.x, a1.y, h, l); Ah.u[2] = h; Al.u[2] = l;
            cvt2(a1.z, a1.w, h, l); Ah.u[3] = h; Al.u[3] = l;
            cvt2(a2.x, a2.y, h, l); Ah.u[4] = h; Al.u[4] = l;
            cvt2(a2.z, a2.w, h, l); Ah.u[5] = h; Al.u[5] = l;
            cvt2(a3.x, a3.y, h, l); Ah.u[6] = h; Al.u[6] = l;
            cvt2(a3.z, a3.w, h, l); Ah.u[7] = h; Al.u[7] = l;
        }
#pragma unroll
        for (int nb = 0; nb < 8; nb += 2) {
            const int rb0 = (nb << 4) + l15;
            const int rb1 = rb0 + 16;
            FragU Bh0, Bl0, Bh1, Bl1;
            load_frag(&sWh[rb0 * LDSK + k0], Bh0);
            load_frag(&sWl[rb0 * LDSK + k0], Bl0);
            load_frag(&sWh[rb1 * LDSK + k0], Bh1);
            load_frag(&sWl[rb1 * LDSK + k0], Bl1);
            WMMA3X2(acc[nb], acc[nb + 1], Ah, Al, Bh0, Bl0, Bh1, Bl1);
        }
    }

    softpool_reduce(acc, &red, tid, wave, lane, l15, halfsel, dense, p);
}

// label-smoothed symmetric cross-entropy on the 64x64 dense-similarity matrix
__global__ __launch_bounds__(64) void clip_loss_kernel(const float* __restrict__ dense,
                                                       const float* __restrict__ scale_p,
                                                       float* __restrict__ out) {
    __shared__ float redbuf[64];
    const int i = threadIdx.x;
    const float scale = fminf(scale_p[0], 40.0f);
    float part = 0.f;
    const float vii = scale * dense[i * 64 + i];
    {
        float m = -1e30f;
        for (int j = 0; j < 64; ++j) m = fmaxf(m, scale * dense[i * 64 + j]);
        float z = 0.f, sum = 0.f;
        for (int j = 0; j < 64; ++j) {
            const float v = scale * dense[i * 64 + j];
            z += __expf(v - m); sum += v;
        }
        const float lse = m + __logf(z);
        part += 0.9f * (lse - vii) + 0.1f * (lse - sum * (1.0f / 64.0f));
    }
    {
        float m = -1e30f;
        for (int j = 0; j < 64; ++j) m = fmaxf(m, scale * dense[j * 64 + i]);
        float z = 0.f, sum = 0.f;
        for (int j = 0; j < 64; ++j) {
            const float v = scale * dense[j * 64 + i];
            z += __expf(v - m); sum += v;
        }
        const float lse = m + __logf(z);
        part += 0.9f * (lse - vii) + 0.1f * (lse - sum * (1.0f / 64.0f));
    }
    redbuf[i] = part;
    __syncthreads();
    if (i == 0) {
        float t = 0.f;
        for (int k = 0; k < 64; ++k) t += redbuf[k];
        out[0] = t * (1.0f / 128.0f);   // /64 (batch mean) /2 (two losses)
    }
}

extern "C" void kernel_launch(void* const* d_in, const int* in_sizes, int n_in,
                              void* d_out, int out_size, void* d_ws, size_t ws_size,
                              hipStream_t stream) {
    const float* V  = (const float*)d_in[0];   // [64,128,256] f32
    const float* W  = (const float*)d_in[1];   // [64,128,256] f32
    const float* ls = (const float*)d_in[2];   // scalar logit_scale
    float* dense = (float*)d_ws;               // 64*64 f32 at ws offset 0

    const size_t N    = (size_t)64 * 128 * 256;
    const size_t need = 16384 + 4 * N * sizeof(unsigned short);   // dense + Vh/Vl/Wh/Wl

    if (ws_size >= need) {
        unsigned short* Vh = (unsigned short*)((char*)d_ws + 16384);
        unsigned short* Vl = Vh + N;
        unsigned short* Wh = Vl + N;
        unsigned short* Wl = Wh + N;
        precvt_kernel<<<dim3(4096), dim3(256), 0, stream>>>(V, W, Vh, Vl, Wh, Wl);
        clip_pair_pre<<<dim3(4096), dim3(256), 0, stream>>>(Vh, Vl, Wh, Wl, dense);
    } else {
        clip_pair_f32<<<dim3(4096), dim3(256), 0, stream>>>(V, W, dense);
    }
    clip_loss_kernel<<<dim3(1), dim3(64), 0, stream>>>(dense, ls, (float*)d_out);
}
